// AttentionPool_2448131359412
// MI455X (gfx1250) — matrix operand order repaired and checked by reading, not compiled, not run on hardware
//
#include <hip/hip_runtime.h>
#include <hip/hip_bf16.h>

typedef __attribute__((ext_vector_type(16))) _Float16 v16h;
typedef __attribute__((ext_vector_type(8)))  float    v8f;

#define D_DIM   768
#define S_DIM   4096
#define B_DIM   8
#define ROWS    (B_DIM * S_DIM)          // 32768
#define KT      24                       // 768 / 32
#define NT      48                       // 768 / 16
#define WPAD    776                      // 768 + 8 halfs (bank spread)
#define SCHUNKS 64                       // s-chunks for pooling

// ---------------- tanh: prefer v_tanh_f32 (gfx1250 TRANS op) ---------------
__device__ __forceinline__ float fast_tanh(float x) {
#if __has_builtin(__builtin_amdgcn_tanhf)
    return __builtin_amdgcn_tanhf(x);
#else
    x = fminf(fmaxf(x, -10.0f), 10.0f);
    float e = __expf(2.0f * x);
    return (e - 1.0f) / (e + 1.0f);
#endif
}

// ---------------- K0: W [k][n] fp32 -> WT [n][k] fp16 (LDS-tiled) ----------
__global__ void AttentionPool_transposeW(const float* __restrict__ W,
                                         _Float16* __restrict__ WT) {
    __shared__ float tile[32][33];
    int bx = blockIdx.x % (D_DIM / 32);   // n tile
    int by = blockIdx.x / (D_DIM / 32);   // k tile
    int c  = threadIdx.x & 31;
    int r0 = threadIdx.x >> 5;            // 0..7
#pragma unroll
    for (int i = 0; i < 4; ++i) {
        int r = r0 + i * 8;
        tile[r][c] = W[(size_t)(by * 32 + r) * D_DIM + bx * 32 + c];
    }
    __syncthreads();
#pragma unroll
    for (int i = 0; i < 4; ++i) {
        int r = r0 + i * 8;
        WT[(size_t)(bx * 32 + r) * D_DIM + by * 32 + c] = (_Float16)tile[c][r];
    }
}

// ---- async panel staging: 24KB panel, 16B per lane per op, ASYNCcnt -------
__device__ __forceinline__ void stage_async(const _Float16* __restrict__ src,
                                            _Float16* lds_base, int tid) {
#pragma unroll
    for (int i = 0; i < 6; ++i) {
        const int u  = tid + i * 256;            // 1536 units of 8 halfs
        const int n  = u / (D_DIM / 8);
        const int ku = (u % (D_DIM / 8)) * 8;
        const _Float16* g = src + (size_t)n * D_DIM + ku;
        const unsigned lds = (unsigned)(unsigned long long)(lds_base + n * WPAD + ku);
        asm volatile("global_load_async_to_lds_b128 %0, %1, off"
                     :: "v"(lds), "v"(g) : "memory");
    }
}

// ---------------- K1: fused GEMM + tanh + dot(v) -> scores -----------------
// block = 256 threads (8 waves); wave w owns rows blockIdx*128 + w*16 .. +16
__global__ void __launch_bounds__(256)
AttentionPool_scores(const float* __restrict__ x,
                     const _Float16* __restrict__ WT,
                     const float* __restrict__ vvec,
                     float* __restrict__ scores) {
    __shared__ _Float16 wp[2][16 * WPAD];   // double-buffered 768x16 panel

    const int lane = threadIdx.x & 31;
    const int wave = threadIdx.x >> 5;
    const bool hi  = lane >= 16;
    const int nloc = lane & 15;
    const int rowbase = blockIdx.x * 128 + wave * 16;
    const int row     = rowbase + nloc;

    // ---- load all A fragments (16 rows x 768 K) once, fp32 -> fp16 ----
    // A 16x32 layout: lanes 0-15 hold K = kb..kb+7 and kb+16..kb+23 (kb = kt*32)
    //                 lanes 16-31 the same shifted by +8.
    const float* xrow = x + (size_t)row * D_DIM;
    v16h afrag[KT];
#pragma unroll
    for (int kt = 0; kt < KT; ++kt) {
        const int kb = kt * 32 + (hi ? 8 : 0);
        const float* p0 = xrow + kb;
        const float* p1 = xrow + kb + 16;
#pragma unroll
        for (int j = 0; j < 8; ++j) {
            afrag[kt][j]     = (_Float16)p0[j];
            afrag[kt][8 + j] = (_Float16)p1[j];
        }
    }

    float rowAcc[8];
#pragma unroll
    for (int r = 0; r < 8; ++r) rowAcc[r] = 0.0f;

    stage_async(WT, &wp[0][0], threadIdx.x);

#pragma unroll 1
    for (int nt = 0; nt < NT; ++nt) {
        // my async writes into the upcoming panel must be done before signaling
        asm volatile("s_wait_asynccnt 0x0" ::: "memory");
        __syncthreads();                       // panel (nt&1) ready for all waves
        if (nt + 1 < NT)
            stage_async(WT + (size_t)(nt + 1) * 16 * D_DIM,
                        &wp[(nt + 1) & 1][0], threadIdx.x);
        const int buf = nt & 1;

        // two independent accumulator chains (even/odd k-tiles) for ILP
        v8f c0, c1;
#pragma unroll
        for (int r = 0; r < 8; ++r) { c0[r] = 0.0f; c1[r] = 0.0f; }

        // B 32x16 layout: lane (N = nloc) holds K = kb..kb+7 and kb+16..kb+23,
        // lanes 16-31 the same shifted by +8 (matches A).
        const _Float16* brow = &wp[buf][nloc * WPAD + (hi ? 8 : 0)];
#pragma unroll
        for (int kt = 0; kt < KT; kt += 2) {
            union { v16h v; uint4 q[2]; } bfa, bfb;
            const _Float16* bp0 = brow + kt * 32;
            const _Float16* bp1 = brow + (kt + 1) * 32;
            bfa.q[0] = *(const uint4*)(bp0);
            bfa.q[1] = *(const uint4*)(bp0 + 16);
            bfb.q[0] = *(const uint4*)(bp1);
            bfb.q[1] = *(const uint4*)(bp1 + 16);
            c0 = __builtin_amdgcn_wmma_f32_16x16x32_f16(
                     false, afrag[kt],     false, bfa.v, (short)0, c0, false, false);
            c1 = __builtin_amdgcn_wmma_f32_16x16x32_f16(
                     false, afrag[kt + 1], false, bfb.v, (short)0, c1, false, false);
        }

        // epilogue: scores += tanh(y) * v[n] ; lane's 8 values share N = nloc
        const float vn = vvec[nt * 16 + nloc];
#pragma unroll
        for (int r = 0; r < 8; ++r)
            rowAcc[r] += fast_tanh(c0[r] + c1[r]) * vn;
    }

    // C layout: lanes 0-15 -> M = r, lanes 16-31 -> M = 8 + r.
    // Reduce over the 16 N-lanes of each half (xor masks < 16 stay in half).
#pragma unroll
    for (int r = 0; r < 8; ++r) {
        float v = rowAcc[r];
#pragma unroll
        for (int m = 8; m >= 1; m >>= 1) v += __shfl_xor(v, m, 32);
        rowAcc[r] = v;
    }
    if (lane == 0) {
#pragma unroll
        for (int r = 0; r < 8; ++r) scores[rowbase + r] = rowAcc[r];
    } else if (lane == 16) {
#pragma unroll
        for (int r = 0; r < 8; ++r) scores[rowbase + 8 + r] = rowAcc[r];
    }
}

// ---------------- K2: per-batch softmax stats (max, sum) -------------------
__global__ void AttentionPool_stats(const float* __restrict__ scores,
                                    float* __restrict__ stats) {
    __shared__ float red[32];
    const int b = blockIdx.x;
    const float* sc = scores + (size_t)b * S_DIM;

    float m = -3.0e38f;
    for (int i = threadIdx.x; i < S_DIM; i += blockDim.x) m = fmaxf(m, sc[i]);
#pragma unroll
    for (int o = 16; o >= 1; o >>= 1) m = fmaxf(m, __shfl_xor(m, o, 32));
    if ((threadIdx.x & 31) == 0) red[threadIdx.x >> 5] = m;
    __syncthreads();
    if (threadIdx.x < 32) {
        float t = red[threadIdx.x];
#pragma unroll
        for (int o = 16; o >= 1; o >>= 1) t = fmaxf(t, __shfl_xor(t, o, 32));
        if (threadIdx.x == 0) red[0] = t;
    }
    __syncthreads();
    const float mx = red[0];
    __syncthreads();

    float s = 0.0f;
    for (int i = threadIdx.x; i < S_DIM; i += blockDim.x) s += __expf(sc[i] - mx);
#pragma unroll
    for (int o = 16; o >= 1; o >>= 1) s += __shfl_xor(s, o, 32);
    if ((threadIdx.x & 31) == 0) red[threadIdx.x >> 5] = s;
    __syncthreads();
    if (threadIdx.x == 0) {
        float t = 0.0f;
        for (int i = 0; i < 32; ++i) t += red[i];
        stats[2 * b]     = mx;
        stats[2 * b + 1] = t;
    }
}

// ---------------- K3: partial weighted pool over s-chunks ------------------
// grid = B * SCHUNKS blocks, 192 threads, each thread owns 4 d columns.
__global__ void AttentionPool_poolPartial(const float* __restrict__ x,
                                          const float* __restrict__ scores,
                                          const float* __restrict__ stats,
                                          float* __restrict__ partials) {
    const int b  = blockIdx.x >> 6;
    const int sc = blockIdx.x & 63;
    const int d4 = threadIdx.x * 4;
    const float mx  = stats[2 * b];
    const float inv = 1.0f / stats[2 * b + 1];

    float a0 = 0.f, a1 = 0.f, a2 = 0.f, a3 = 0.f;
    const int s0 = sc * (S_DIM / SCHUNKS);
    for (int i = 0; i < S_DIM / SCHUNKS; ++i) {
        const int s = s0 + i;
        const float w = __expf(scores[(size_t)b * S_DIM + s] - mx) * inv;
        const float4 xv = *(const float4*)(x + ((size_t)b * S_DIM + s) * D_DIM + d4);
        a0 += w * xv.x; a1 += w * xv.y; a2 += w * xv.z; a3 += w * xv.w;
    }
    float4 r; r.x = a0; r.y = a1; r.z = a2; r.w = a3;
    *(float4*)(partials + ((size_t)(b * SCHUNKS + sc)) * D_DIM + d4) = r;
}

// ---------------- K4: deterministic reduction of partials ------------------
__global__ void AttentionPool_poolReduce(const float* __restrict__ partials,
                                         float* __restrict__ out) {
    const int idx = blockIdx.x * 256 + threadIdx.x;     // < B*D
    if (idx >= B_DIM * D_DIM) return;
    const int b = idx / D_DIM;
    const int d = idx % D_DIM;
    float s = 0.0f;
    for (int c = 0; c < SCHUNKS; ++c)
        s += partials[((size_t)(b * SCHUNKS + c)) * D_DIM + d];
    out[idx] = s;
}

extern "C" void kernel_launch(void* const* d_in, const int* in_sizes, int n_in,
                              void* d_out, int out_size, void* d_ws, size_t ws_size,
                              hipStream_t stream) {
    const float* x     = (const float*)d_in[0];   // [8,4096,768]
    const float* att_v = (const float*)d_in[1];   // [768]
    const float* att_W = (const float*)d_in[2];   // [768,768]
    float* out = (float*)d_out;                   // [8,768]

    char* ws = (char*)d_ws;
    _Float16* WT    = (_Float16*)ws;                       // 768*768*2     = 1,179,648 B
    float*    scores = (float*)(ws + 1179648);             // 32768*4       =   131,072 B
    float*    stats  = (float*)(ws + 1310720);             // 16*4          =        64 B
    float*    parts  = (float*)(ws + 1310784);             // 8*64*768*4    = 1,572,864 B

    AttentionPool_transposeW <<<(D_DIM/32)*(D_DIM/32), 256, 0, stream>>>(att_W, WT);
    AttentionPool_scores     <<<ROWS / 128, 256, 0, stream>>>(x, WT, att_v, scores);
    AttentionPool_stats      <<<B_DIM, 1024, 0, stream>>>(scores, stats);
    AttentionPool_poolPartial<<<B_DIM * SCHUNKS, 192, 0, stream>>>(x, scores, stats, parts);
    AttentionPool_poolReduce <<<(B_DIM * D_DIM + 255) / 256, 256, 0, stream>>>(parts, out);
}
